// BiLSTM_CRF_76845554860437
// MI455X (gfx1250) — matrix operand order repaired and checked by reading, not compiled, not run on hardware
//
#include <hip/hip_runtime.h>
#include <hip/hip_bf16.h>
#include <math.h>

// ---------------- CDNA5 WMMA types ----------------
typedef __attribute__((ext_vector_type(16))) __bf16 v16bf;
typedef __attribute__((ext_vector_type(8)))  float  v8f;

union U16 { unsigned int u[8]; v16bf v; };

// ---------------- problem constants ----------------
#define S_LEN 4096
#define EDIM  256
#define HDIM  256
#define G4H   1024          // 4*H gate rows
#define TAGS  32
#define START_TAG 30
#define STOP_TAG  31
#define NEGV  (-10000.0f)
#define WROWS 576           // gate rows of w_hh pinned in LDS (576*512B = 288KB)

// fp32 -> bf16, round-to-nearest-even
__device__ __forceinline__ unsigned short f2bf(float f) {
  unsigned int u = __float_as_uint(f);
  u += 0x7FFFu + ((u >> 16) & 1u);
  return (unsigned short)(u >> 16);
}
__device__ __forceinline__ float bf2f(unsigned int hbits) {
  return __uint_as_float(hbits << 16);
}
__device__ __forceinline__ float sigm(float x) {
  return 1.0f / (1.0f + __expf(-x));
}

// ================= K0: gather + precision conversion =================
// embbf[s][e] = bf16(embedding[sentence[s]][e])
// wihbf[dir][r][k], whhbf[dir][r][k], linwbf[t][k]  (row-major bf16)
__global__ void k_prep(const int* __restrict__ sent,
                       const float* __restrict__ emb,
                       const float* __restrict__ wihf, const float* __restrict__ wihb,
                       const float* __restrict__ whhf, const float* __restrict__ whhb,
                       const float* __restrict__ linw,
                       unsigned short* __restrict__ embbf,
                       unsigned short* __restrict__ wihbf,
                       unsigned short* __restrict__ whhbf,
                       unsigned short* __restrict__ linwbf) {
  const int N0 = S_LEN * EDIM;          // 1048576
  const int N1 = G4H * EDIM;            // 262144 (per direction)
  const int N3 = TAGS * 2 * HDIM;       // 16384
  const int NT = N0 + 2 * N1 + 2 * N1 + N3;
  for (int i = blockIdx.x * blockDim.x + threadIdx.x; i < NT;
       i += gridDim.x * blockDim.x) {
    int j = i;
    if (j < N0) {
      int s = j >> 8, e = j & 255;
      embbf[j] = f2bf(emb[(size_t)sent[s] * EDIM + e]);
    } else if ((j -= N0) < 2 * N1) {
      const float* src = (j < N1) ? wihf : wihb;
      int jj = (j < N1) ? j : (j - N1);
      wihbf[j] = f2bf(src[jj]);
    } else if ((j -= 2 * N1) < 2 * N1) {
      const float* src = (j < N1) ? whhf : whhb;
      int jj = (j < N1) ? j : (j - N1);
      whhbf[j] = f2bf(src[jj]);
    } else {
      j -= 2 * N1;
      linwbf[j] = f2bf(linw[j]);
    }
  }
}

// ================= K1: xg = emb @ w_ih^T + b (both dirs, WMMA bf16) =========
// xg[dir][u][r]; dir1 time axis is reversed (u corresponds to emb[S-1-u]).
// One wave computes one 16x16 output tile; K=256 in 8 chunks of 32.
__global__ void k_xg(const unsigned short* __restrict__ embbf,
                     const unsigned short* __restrict__ wihbf,
                     const float* __restrict__ b_f, const float* __restrict__ b_b,
                     float* __restrict__ xg) {
  int wave = threadIdx.x >> 5, lane = threadIdx.x & 31;
  int tile = blockIdx.x * 4 + wave;          // 2 * 256 * 64 tiles
  int dir = tile >> 14;                      // 16384 tiles per direction
  int rem = tile & 16383;
  int st = rem >> 6, nt = rem & 63;
  int s0 = st << 4, n0 = nt << 4;
  int mlane = lane & 15, hi = lane >> 4;
  int kbA = hi * 8, kbB = hi * 16;

  int sm = s0 + mlane;
  int sr = dir ? (S_LEN - 1 - sm) : sm;      // reversed sequence for backward dir
  const unsigned short* arow = embbf + (size_t)sr * EDIM;
  const unsigned short* brow = wihbf + (size_t)dir * G4H * EDIM
                                     + (size_t)(n0 + mlane) * EDIM;
  v8f c = {0.f, 0.f, 0.f, 0.f, 0.f, 0.f, 0.f, 0.f};
#pragma unroll
  for (int kc = 0; kc < 8; ++kc) {
    int k0 = kc * 32;
    U16 a, b;
#pragma unroll
    for (int v = 0; v < 4; ++v) {
      a.u[v]     = *(const unsigned int*)(arow + k0 + kbA + 2 * v);
      a.u[4 + v] = *(const unsigned int*)(arow + k0 + 16 + kbA + 2 * v);
    }
#pragma unroll
    for (int v = 0; v < 4; ++v) {
      b.u[v]     = *(const unsigned int*)(brow + k0 + kbA + 2 * v);
      b.u[4 + v] = *(const unsigned int*)(brow + k0 + 16 + kbA + 2 * v);
    }
    c = __builtin_amdgcn_wmma_f32_16x16x32_bf16(false, a.v, false, b.v,
                                                (short)0, c, false, false);
  }
  int n = n0 + mlane;
  float bias = (dir ? b_b : b_f)[n];
  float* xgd = xg + (size_t)dir * S_LEN * G4H;
#pragma unroll
  for (int r = 0; r < 8; ++r) {
    int m = r + 8 * hi;
    xgd[(size_t)(s0 + m) * G4H + n] = c[r] + bias;
  }
}

// ================= K2: persistent per-direction LSTM recurrence =============
// One 1024-thread workgroup per direction; thread t owns gate row t.
// 576 rows of w_hh (bf16) pinned in LDS via global_load_async_to_lds_b128
// (ASYNCcnt domain, VGPR-bypassing); remaining 448 rows streamed from L2.
__global__ void k_lstm(const float* __restrict__ xg,
                       const unsigned short* __restrict__ whhbf,
                       const float* __restrict__ h0, const float* __restrict__ c0,
                       float* __restrict__ hs) {
  extern __shared__ char smem[];
  unsigned int* wlds = (unsigned int*)smem;                 // WROWS*128 dwords
  float* gates = (float*)(smem + (size_t)WROWS * 512);      // 1024 floats
  float* hbuf  = gates + G4H;                               // 256 floats

  const int dir = blockIdx.x;
  const int t = threadIdx.x;
  const unsigned int* whh = (const unsigned int*)whhbf + (size_t)dir * G4H * 128;

  // ---- stage LDS-resident weight rows with async direct-to-LDS copies ----
  // 288KB = 18432 x b128; 18 issues per thread. 16B-aligned on both sides.
  {
    unsigned lds_base = (unsigned)(uintptr_t)smem;           // LDS aperture low bits
    const char* gsrc = (const char*)whh;
    for (int i = t; i < WROWS * 32; i += 1024) {
      unsigned lo = lds_base + (unsigned)(i * 16);
      unsigned long long ga = (unsigned long long)(uintptr_t)(gsrc + (size_t)i * 16);
      asm volatile("global_load_async_to_lds_b128 %0, %1, off"
                   :: "v"(lo), "v"(ga) : "memory");
    }
    asm volatile("s_wait_asynccnt 0" ::: "memory");
  }

  float creg = 0.0f;
  if (t < HDIM) {
    creg = c0[dir * HDIM + t];
    hbuf[t] = h0[dir * HDIM + t];
  }
  __syncthreads();

  const unsigned int* wr = (t < WROWS) ? (wlds + (size_t)t * 128)
                                       : (whh + (size_t)t * 128);
  const float* xgd = xg + (size_t)dir * S_LEN * G4H;
  float* hsd = hs + (size_t)dir * S_LEN * HDIM;

  for (int s = 0; s < S_LEN; ++s) {
    float acc = xgd[(size_t)s * G4H + t];
#pragma unroll 16
    for (int k2 = 0; k2 < 128; ++k2) {
      unsigned int p = wr[k2];                 // two bf16 weights
      acc = fmaf(bf2f(p & 0xFFFFu), hbuf[2 * k2],     acc);
      acc = fmaf(bf2f(p >> 16),     hbuf[2 * k2 + 1], acc);
    }
    gates[t] = acc;
    if (s + 1 < S_LEN) __builtin_prefetch(&xgd[(size_t)(s + 1) * G4H + t], 0, 1);
    __syncthreads();
    if (t < HDIM) {
      float ig = sigm(gates[t]);
      float fg = sigm(gates[HDIM + t]);
      float gg = tanhf(gates[2 * HDIM + t]);
      float og = sigm(gates[3 * HDIM + t]);
      creg = fg * creg + ig * gg;
      float hh = og * tanhf(creg);
      hsd[(size_t)s * HDIM + t] = hh;
      hbuf[t] = hh;
    }
    __syncthreads();
  }
}

// ================= K3: feats = [hs_f, hs_b] @ lin_w^T + lin_b (WMMA) ========
__global__ void k_feats(const float* __restrict__ hs,
                        const unsigned short* __restrict__ linwbf,
                        const float* __restrict__ lin_b,
                        float* __restrict__ feats) {
  int wave = threadIdx.x >> 5, lane = threadIdx.x & 31;
  int tile = blockIdx.x * 4 + wave;          // 256 * 2 tiles
  int st = tile >> 1, nt = tile & 1;
  int s0 = st << 4, n0 = nt << 4;
  int mlane = lane & 15, hi = lane >> 4;
  int kbA = hi * 8, kbB = hi * 16;
  int sm = s0 + mlane;

  const float* hsf = hs;
  const float* hsb = hs + (size_t)S_LEN * HDIM;     // backward dir, reversed time
  const unsigned short* brow = linwbf + (size_t)(n0 + mlane) * (2 * HDIM);

  v8f c = {0.f, 0.f, 0.f, 0.f, 0.f, 0.f, 0.f, 0.f};
#pragma unroll
  for (int kc = 0; kc < 16; ++kc) {
    int k0 = kc * 32;
    U16 a, b;
#pragma unroll
    for (int v = 0; v < 8; ++v) {
      int kA = (v < 4) ? (k0 + kbA + 2 * v) : (k0 + 16 + kbA + 2 * (v - 4));
      float f0, f1;
      if (kA < HDIM) {
        f0 = hsf[(size_t)sm * HDIM + kA];
        f1 = hsf[(size_t)sm * HDIM + kA + 1];
      } else {
        size_t u = (size_t)(S_LEN - 1 - sm);   // un-reverse backward hidden states
        f0 = hsb[u * HDIM + (kA - HDIM)];
        f1 = hsb[u * HDIM + (kA - HDIM + 1)];
      }
      a.u[v] = (unsigned int)f2bf(f0) | ((unsigned int)f2bf(f1) << 16);
      b.u[v] = *(const unsigned int*)(brow + kA);
    }
    c = __builtin_amdgcn_wmma_f32_16x16x32_bf16(false, a.v, false, b.v,
                                                (short)0, c, false, false);
  }
  int n = n0 + mlane;
  float bias = lin_b[n];
#pragma unroll
  for (int r = 0; r < 8; ++r)
    feats[(size_t)(s0 + r + 8 * hi) * TAGS + n] = c[r] + bias;
}

// ================= K4: Viterbi forward + LDS backtrack ======================
// Single wave; lane n = next tag. Backpointers kept as bytes in LDS (128KB).
__global__ void k_viterbi(const float* __restrict__ feats,
                          const float* __restrict__ trans,
                          float* __restrict__ out) {
  extern __shared__ char smem[];
  unsigned char* bp = (unsigned char*)smem;               // S_LEN*TAGS bytes
  float* fv = (float*)(smem + (size_t)S_LEN * TAGS);      // TAGS floats
  const int n = threadIdx.x;

  float trow[TAGS];
#pragma unroll
  for (int p = 0; p < TAGS; ++p) trow[p] = trans[n * TAGS + p];
  fv[n] = (n == START_TAG) ? 0.0f : NEGV;
  __syncthreads();

  for (int s = 0; s < S_LEN; ++s) {
    float best = -3.4e38f;
    int bi = 0;
#pragma unroll
    for (int p = 0; p < TAGS; ++p) {
      float sc = fv[p] + trow[p];
      if (sc > best) { best = sc; bi = p; }   // strict > == argmax first index
    }
    float fn = best + feats[(size_t)s * TAGS + n];
    bp[(size_t)s * TAGS + n] = (unsigned char)bi;
    __syncthreads();
    fv[n] = fn;
    __syncthreads();
  }

  if (n == 0) {
    float best = -3.4e38f;
    int last = 0;
    for (int p = 0; p < TAGS; ++p) {
      float sc = fv[p] + trans[STOP_TAG * TAGS + p];
      if (sc > best) { best = sc; last = p; }
    }
    out[0] = best;
    int idx = last;
    for (int s = S_LEN - 1; s >= 0; --s) {
      out[1 + s] = (float)idx;
      idx = bp[(size_t)s * TAGS + idx];
    }
  }
}

// ================= host-side orchestration ==================================
extern "C" void kernel_launch(void* const* d_in, const int* in_sizes, int n_in,
                              void* d_out, int out_size, void* d_ws, size_t ws_size,
                              hipStream_t stream) {
  const int*   sent = (const int*)d_in[0];
  const float* h0   = (const float*)d_in[1];
  const float* c0   = (const float*)d_in[2];
  const float* emb  = (const float*)d_in[3];
  const float* wihf = (const float*)d_in[4];
  const float* whhf = (const float*)d_in[5];
  const float* b_f  = (const float*)d_in[6];
  const float* wihb = (const float*)d_in[7];
  const float* whhb = (const float*)d_in[8];
  const float* b_b  = (const float*)d_in[9];
  const float* linw = (const float*)d_in[10];
  const float* linb = (const float*)d_in[11];
  const float* trans = (const float*)d_in[12];
  float* out = (float*)d_out;

  // workspace layout (256B aligned)
  char* ws = (char*)d_ws;
  size_t o = 0;
  auto take = [&](size_t bytes) {
    size_t r = o;
    o = (o + bytes + 255) & ~(size_t)255;
    return r;
  };
  unsigned short* embbf  = (unsigned short*)(ws + take((size_t)S_LEN * EDIM * 2));
  unsigned short* wihbf  = (unsigned short*)(ws + take((size_t)2 * G4H * EDIM * 2));
  unsigned short* whhbf  = (unsigned short*)(ws + take((size_t)2 * G4H * HDIM * 2));
  unsigned short* linwbf = (unsigned short*)(ws + take((size_t)TAGS * 2 * HDIM * 2));
  float* xg    = (float*)(ws + take((size_t)2 * S_LEN * G4H * 4));
  float* hs    = (float*)(ws + take((size_t)2 * S_LEN * HDIM * 4));
  float* feats = (float*)(ws + take((size_t)S_LEN * TAGS * 4));
  (void)ws_size; (void)in_sizes; (void)n_in; (void)out_size;

  const int lstm_lds = WROWS * 512 + G4H * 4 + HDIM * 4;       // ~300KB (<=320KB WGP LDS)
  const int vit_lds  = S_LEN * TAGS + TAGS * 4;                // ~128KB
  (void)hipFuncSetAttribute((const void*)k_lstm,
                            hipFuncAttributeMaxDynamicSharedMemorySize, lstm_lds);
  (void)hipFuncSetAttribute((const void*)k_viterbi,
                            hipFuncAttributeMaxDynamicSharedMemorySize, vit_lds);

  k_prep<<<2048, 256, 0, stream>>>(sent, emb, wihf, wihb, whhf, whhb, linw,
                                   embbf, wihbf, whhbf, linwbf);
  k_xg<<<8192, 128, 0, stream>>>(embbf, wihbf, b_f, b_b, xg);
  k_lstm<<<2, 1024, lstm_lds, stream>>>(xg, whhbf, h0, c0, hs);
  k_feats<<<128, 128, 0, stream>>>(hs, linwbf, linb, feats);
  k_viterbi<<<1, 32, vit_lds, stream>>>(feats, trans, out);
}
